// FeatureExtractor_57784490001190
// MI455X (gfx1250) — matrix operand order repaired and checked by reading, not compiled, not run on hardware
//
#include <hip/hip_runtime.h>
#include <hip/hip_bf16.h>

typedef _Float16 h16;
typedef __attribute__((ext_vector_type(16))) _Float16 v16h;
typedef __attribute__((ext_vector_type(8)))  float    v8f;

// ---------------------------------------------------------------------------
// WMMA fragment helpers (v_wmma_f32_16x16x32_f16, D = A(16x32) * B(32x16) + C)
// A: activations row-major [row][K];  B: weights row-major [outch][K]
// ---------------------------------------------------------------------------
__device__ __forceinline__ v16h frag_a(const h16* __restrict__ act, int ld,
                                       int row0, int k0) {
  const int lane = threadIdx.x & 31;
  const h16* p = act + (size_t)(row0 + (lane & 15)) * ld + k0 + ((lane & 16) ? 8 : 0);
  v16h a;
#pragma unroll
  for (int j = 0; j < 8; ++j) { a[j] = p[j]; a[j + 8] = p[16 + j]; }
  return a;
}

__device__ __forceinline__ v16h frag_b(const h16* __restrict__ w, int ld,
                                       int n0, int k0) {
  const int lane = threadIdx.x & 31;
  const h16* p = w + (size_t)(n0 + (lane & 15)) * ld + k0 + ((lane & 16) ? 8 : 0);
  v16h b;
#pragma unroll
  for (int j = 0; j < 8; ++j) { b[j] = p[j]; b[j + 8] = p[16 + j]; }
  return b;
}

#define WMMA_F16(a, b, c) \
  __builtin_amdgcn_wmma_f32_16x16x32_f16(false, (a), false, (b), (short)0, (c), false, false)

// ---------------------------------------------------------------------------
// Register-resident-B GEMM: one wave holds all K/32 B-frags of its out-tile and
// sweeps ROWS row-tiles.  out[row,col] = act[row,:] . w[col,:] + bias[col]
// ---------------------------------------------------------------------------
template <int K, int ROWS>
__global__ void gemm_res(const h16* __restrict__ act, const h16* __restrict__ w,
                         const float* __restrict__ bias, h16* __restrict__ of16,
                         int nrows, int otiles, int ldo) {
  constexpr int NF = K / 32;
  const int wave = threadIdx.x >> 5;
  const int lane = threadIdx.x & 31;
  const int wg = blockIdx.x * 8 + wave;
  const int total = (nrows / (16 * ROWS)) * otiles;
  if (wg >= total) return;                 // wave-uniform (EXEC all-1 for WMMA)
  const int rg = wg / otiles, ot = wg - rg * otiles;
  const int col = ot * 16 + (lane & 15);
  const float bv = bias ? bias[col] : 0.f;

  v16h bf[NF];
#pragma unroll
  for (int f = 0; f < NF; ++f) bf[f] = frag_b(w, K, ot * 16, f * 32);

  const int hi8 = (lane & 16) ? 8 : 0;
#pragma unroll
  for (int r = 0; r < ROWS; ++r) {
    const int row0 = (rg * ROWS + r) * 16;
    v8f c;
#pragma unroll
    for (int v = 0; v < 8; ++v) c[v] = bv;
#pragma unroll
    for (int f = 0; f < NF; ++f) c = WMMA_F16(frag_a(act, K, row0, f * 32), bf[f], c);
#pragma unroll
    for (int v = 0; v < 8; ++v)
      of16[(size_t)(row0 + v + hi8) * ldo + col] = (h16)c[v];
  }
}

// ---------------------------------------------------------------------------
// Fused qt/kx projections (K=64): A frags loaded once, two resident B sets
// ---------------------------------------------------------------------------
template <int ROWS>
__global__ void gemm_qk(const h16* __restrict__ act, const h16* __restrict__ wt,
                        const h16* __restrict__ wx, float* __restrict__ qt,
                        float* __restrict__ kx, int nrows) {
  const int wave = threadIdx.x >> 5;
  const int lane = threadIdx.x & 31;
  const int wg = blockIdx.x * 8 + wave;
  const int total = (nrows / (16 * ROWS)) * 4;
  if (wg >= total) return;
  const int rg = wg >> 2, ot = wg & 3;
  const int col = ot * 16 + (lane & 15);
  const int hi8 = (lane & 16) ? 8 : 0;

  v16h bt0 = frag_b(wt, 64, ot * 16, 0), bt1 = frag_b(wt, 64, ot * 16, 32);
  v16h bx0 = frag_b(wx, 64, ot * 16, 0), bx1 = frag_b(wx, 64, ot * 16, 32);

#pragma unroll
  for (int r = 0; r < ROWS; ++r) {
    const int row0 = (rg * ROWS + r) * 16;
    v16h a0 = frag_a(act, 64, row0, 0);
    v16h a1 = frag_a(act, 64, row0, 32);
    v8f cq, ck;
#pragma unroll
    for (int v = 0; v < 8; ++v) { cq[v] = 0.f; ck[v] = 0.f; }
    cq = WMMA_F16(a0, bt0, cq);
    cq = WMMA_F16(a1, bt1, cq);
    ck = WMMA_F16(a0, bx0, ck);
    ck = WMMA_F16(a1, bx1, ck);
#pragma unroll
    for (int v = 0; v < 8; ++v) {
      const size_t idx = (size_t)(row0 + v + hi8) * 64 + col;
      qt[idx] = cq[v];
      kx[idx] = ck[v];
    }
  }
}

// ---------------------------------------------------------------------------
// Streaming GEMM (large K, bias, optional relu) — used for m1 (K=960)
// ---------------------------------------------------------------------------
__global__ void gemm16(const h16* __restrict__ act, const h16* __restrict__ w,
                       const float* __restrict__ bias, h16* __restrict__ of16,
                       int nrows, int K, int otiles, int ldo, int relu) {
  const int wave = threadIdx.x >> 5;
  const int lane = threadIdx.x & 31;
  const int wg = blockIdx.x * 8 + wave;
  const int total = (nrows >> 4) * otiles;
  if (wg >= total) return;
  const int pt = wg / otiles, ot = wg - pt * otiles;
  const int n0 = ot * 16, row0 = pt * 16;
  const int col = n0 + (lane & 15);
  const float bv = bias ? bias[col] : 0.f;
  const h16* arow = act + (size_t)(row0 + (lane & 15)) * K;
  v8f c;
#pragma unroll
  for (int v = 0; v < 8; ++v) c[v] = bv;
  for (int k0 = 0; k0 < K; k0 += 32) {
    if (k0 + 32 < K) __builtin_prefetch(arow + k0 + 32, 0, 0);  // global_prefetch_b8
    v16h a = frag_a(act, K, row0, k0);
    v16h b = frag_b(w, K, n0, k0);
    c = WMMA_F16(a, b, c);
  }
  const int rbase = row0 + ((lane & 16) ? 8 : 0);
#pragma unroll
  for (int v = 0; v < 8; ++v) {
    float val = c[v];
    if (relu) val = fmaxf(val, 0.f);
    of16[(size_t)(rbase + v) * ldo + col] = (h16)val;
  }
}

// ---------------------------------------------------------------------------
// Prep 1: collapse the pointwise tower over the 23-symbol alphabet into tables
// ---------------------------------------------------------------------------
__global__ void prep_tables(const float* __restrict__ emb,
    const float* dr_w, const float* dr_b, const float* c1_w, const float* c1_b,
    const float* c11_w, const float* c11_b, const float* c12_w, const float* c12_b,
    const float* c21_w, const float* c21_b, const float* c22_w, const float* c22_b,
    const float* c31_w, const float* c31_b, const float* c32_w, const float* c32_b,
    const float* c41_w, const float* c41_b, const float* fc_w, const float* fc_b,
    float* __restrict__ table_r, float* __restrict__ table_acc,
    float* __restrict__ bias_h, h16* __restrict__ WM) {
  __shared__ float e[128], r0[64], r[64], t1[64], x2[64], x3[64], x4[64];
  const int o = threadIdx.x;   // 64 threads
  for (int v = 0; v < 23; ++v) {
    e[o] = emb[v * 128 + o]; e[o + 64] = emb[v * 128 + 64 + o];
    __syncthreads();
    float s = dr_b[o];
    for (int k = 0; k < 128; ++k) s += dr_w[o * 128 + k] * e[k];
    r0[o] = s; __syncthreads();
    s = c1_b[o];
    for (int k = 0; k < 64; ++k) s += c1_w[o * 64 + k] * r0[k];
    r[o] = s; table_r[v * 64 + o] = s; __syncthreads();
    s = c11_b[o];
    for (int k = 0; k < 64; ++k) s += c11_w[o * 64 + k] * r[k];
    t1[o] = s; __syncthreads();
    s = c12_b[o];
    for (int k = 0; k < 64; ++k) s += c12_w[o * 64 + k] * t1[k];
    x2[o] = s; __syncthreads();
    s = c21_b[o];
    for (int k = 0; k < 64; ++k) s += c21_w[o * 64 + k] * r[k];
    t1[o] = s; __syncthreads();
    s = c22_b[o];
    for (int k = 0; k < 64; ++k) s += c22_w[o * 64 + k] * t1[k];
    x3[o] = s; __syncthreads();
    s = c31_b[o];
    for (int k = 0; k < 64; ++k) s += c31_w[o * 64 + k] * r[k];
    t1[o] = s; __syncthreads();
    s = c32_b[o];
    for (int k = 0; k < 64; ++k) s += c32_w[o * 64 + k] * t1[k];
    x4[o] = s; __syncthreads();
    float acc = 0.f;
    for (int k = 0; k < 64; ++k) {
      acc += fc_w[o * 320 + k] * r[k];
      acc += fc_w[o * 320 + 64 + k] * x2[k];
      acc += fc_w[o * 320 + 128 + k] * x3[k];
      acc += fc_w[o * 320 + 192 + k] * x4[k];
    }
    table_acc[v * 64 + o] = acc;
    __syncthreads();
  }
  float s = fc_b[o];
  for (int k = 0; k < 64; ++k) s += fc_w[o * 320 + 256 + k] * c41_b[k];
  bias_h[o] = s;
  for (int k2 = 0; k2 < 64; ++k2) {
    float m = 0.f;
    for (int k = 0; k < 64; ++k) m += fc_w[o * 320 + 256 + k] * c41_w[k * 64 + k2];
    WM[o * 64 + k2] = (h16)m;
  }
}

// ---------------------------------------------------------------------------
// Prep 2: reindex conv/fc weights into f16 row-major [outch][K] WMMA-B layouts
// ---------------------------------------------------------------------------
__global__ void prep_weights(const float* __restrict__ cp1_w,
    const float* __restrict__ cp2_w, const float* __restrict__ wt,
    const float* __restrict__ wx, const float* __restrict__ m1_w,
    const float* __restrict__ m1_b, h16* W1f, h16* W2f, h16* Wtf, h16* Wxf,
    h16* M1f, float* b100) {
  const int tid = blockIdx.x * blockDim.x + threadIdx.x;
  const int stride = gridDim.x * blockDim.x;
  for (int i = tid; i < 64 * 256; i += stride) {       // cp1 [64,64,4]->[o][k*64+c]
    int o = i >> 8, kc = i & 255, k = kc >> 6, c = kc & 63;
    W1f[i] = (h16)cp1_w[o * 256 + c * 4 + k];
  }
  for (int i = tid; i < 64 * 320; i += stride) {       // cp2 [64,64,5]->[o][k*64+c]
    int o = i / 320, kc = i - o * 320, k = kc >> 6, c = kc & 63;
    W2f[i] = (h16)cp2_w[o * 320 + c * 5 + k];
  }
  for (int i = tid; i < 64 * 64; i += stride) {
    Wtf[i] = (h16)wt[i]; Wxf[i] = (h16)wx[i];
  }
  for (int i = tid; i < 112 * 960; i += stride) {      // m1 -> padded [112][t*64+c]
    int o = i / 960, kc = i - o * 960, t = kc >> 6, c = kc & 63;
    M1f[i] = (o < 100) ? (h16)m1_w[o * 960 + c * 15 + t] : (h16)0.f;
  }
  for (int i = tid; i < 112; i += stride) b100[i] = (i < 100) ? m1_b[i] : 0.f;
}

// ---------------------------------------------------------------------------
// Stage 1: H[p,:] = table_acc[x[p]] + M @ maxpool3(table_r[x[p-1..p+1]]) + bias_h
// ---------------------------------------------------------------------------
__global__ void stage1_h(const int* __restrict__ x,
                         const float* __restrict__ table_r,
                         const float* __restrict__ table_acc,
                         const float* __restrict__ bias_h,
                         const h16* __restrict__ WM, h16* __restrict__ H) {
  __shared__ float tr[23 * 64];
  __shared__ h16 pm[64 * 64];
  const int tid = threadIdx.x;
  const int p0 = blockIdx.x * 64;
  for (int i = tid; i < 23 * 64; i += 256) tr[i] = table_r[i];
  __syncthreads();
  for (int idx = tid; idx < 64 * 64; idx += 256) {
    const int pos = idx >> 6, ch = idx & 63;
    const int p = p0 + pos, l = p % 300;
    float m = tr[x[p] * 64 + ch];
    if (l > 0)   m = fmaxf(m, tr[x[p - 1] * 64 + ch]);
    if (l < 299) m = fmaxf(m, tr[x[p + 1] * 64 + ch]);
    pm[pos * 64 + ch] = (h16)m;
  }
  __syncthreads();
  const int wave = tid >> 5, lane = tid & 31;
  const int pt = wave & 3;
  const int hi = (lane & 16) ? 1 : 0;
#pragma unroll
  for (int t = 0; t < 2; ++t) {
    const int ot = (wave >> 2) * 2 + t;
    const int n = ot * 16 + (lane & 15);
    v8f c;
#pragma unroll
    for (int v = 0; v < 8; ++v) {
      const int gp = p0 + pt * 16 + v + hi * 8;
      c[v] = table_acc[x[gp] * 64 + n] + bias_h[n];
    }
#pragma unroll
    for (int k0 = 0; k0 < 64; k0 += 32) {
      const h16* pa = pm + (pt * 16 + (lane & 15)) * 64 + k0 + (hi ? 8 : 0);
      v16h a;
#pragma unroll
      for (int j = 0; j < 8; ++j) { a[j] = pa[j]; a[j + 8] = pa[16 + j]; }
      v16h b = frag_b(WM, 64, ot * 16, k0);
      c = WMMA_F16(a, b, c);
    }
#pragma unroll
    for (int v = 0; v < 8; ++v) {
      const int gp = p0 + pt * 16 + v + hi * 8;
      H[(size_t)gp * 64 + n] = (h16)c[v];
    }
  }
}

// ---------------------------------------------------------------------------
// Stage 4: additive attention per batch.
// scores: e[i,j] = sum_d wa[d]*tanh(qt[i,d]+kx[j,d]+bh[d]) + ba   (trans pipe)
// softmax rows -> LDS a_s[80x96] f16 (zero padded), then
// context: V = relu(a_s @ ain)   via 16x16x32 WMMA tiles
// ---------------------------------------------------------------------------
__global__ void attention(const float* __restrict__ qt,
                          const float* __restrict__ kx,
                          const h16* __restrict__ ain,
                          const float* __restrict__ bh,
                          const float* __restrict__ wa,
                          const float* __restrict__ ba, h16* __restrict__ V) {
  __shared__ h16 qs[75 * 64];
  __shared__ h16 ks[75 * 64];
  __shared__ h16 as_[96 * 64];   // a_in, rows 75..95 zeroed (K padding for WMMA)
  __shared__ h16 a_s[80 * 96];   // softmax weights, padded i->80, j->96
  __shared__ float was[64], bhs[64];
  const int b = blockIdx.x;
  const int tid = threadIdx.x;
  const size_t base = (size_t)b * 75 * 64;
  for (int i = tid; i < 75 * 64; i += 256) {
    qs[i] = (h16)qt[base + i]; ks[i] = (h16)kx[base + i]; as_[i] = ain[base + i];
  }
  for (int i = tid; i < 21 * 64; i += 256) as_[75 * 64 + i] = (h16)0.f;
  for (int i = tid; i < 80 * 96; i += 256) a_s[i] = (h16)0.f;
  if (tid < 64) { was[tid] = wa[tid]; bhs[tid] = bh[tid]; }
  __syncthreads();
  const float bav = ba[0];
  const int wave = tid >> 5, lane = tid & 31;
  for (int i = wave; i < 75; i += 8) {
    float ev[3];
#pragma unroll
    for (int t = 0; t < 3; ++t) {
      const int j = lane + t * 32;
      if (j < 75) {
        float s = 0.f;
        for (int d = 0; d < 64; ++d)
          s += was[d] * tanhf((float)qs[i * 64 + d] + (float)ks[j * 64 + d] + bhs[d]);
        ev[t] = s + bav;
      } else {
        ev[t] = -3.4e38f;
      }
    }
    float m = fmaxf(fmaxf(ev[0], ev[1]), ev[2]);
    for (int off = 16; off; off >>= 1) m = fmaxf(m, __shfl_xor(m, off, 32));
    float a0 = __expf(ev[0] - m);
    float a1 = __expf(ev[1] - m);
    float a2 = (lane < 11) ? __expf(ev[2] - m) : 0.f;
    float s = a0 + a1 + a2;
    for (int off = 16; off; off >>= 1) s += __shfl_xor(s, off, 32);
    const float inv = 1.f / s;
    a_s[i * 96 + lane]      = (h16)(a0 * inv);
    a_s[i * 96 + lane + 32] = (h16)(a1 * inv);
    if (lane < 11) a_s[i * 96 + lane + 64] = (h16)(a2 * inv);
  }
  __syncthreads();
  // context matmul: [80x96] @ [96x64] in 16x16 tiles (5 i-tiles x 4 d-tiles)
  const int hi = (lane & 16) ? 1 : 0;
  for (int t = wave; t < 20; t += 8) {
    const int it = t >> 2, dt = t & 3;
    const int d = dt * 16 + (lane & 15);
    v8f c;
#pragma unroll
    for (int v = 0; v < 8; ++v) c[v] = 0.f;
#pragma unroll
    for (int j0 = 0; j0 < 96; j0 += 32) {
      const h16* pa = a_s + (it * 16 + (lane & 15)) * 96 + j0 + (hi ? 8 : 0);
      v16h a;
#pragma unroll
      for (int j = 0; j < 8; ++j) { a[j] = pa[j]; a[j + 8] = pa[16 + j]; }
      const int jb = j0 + (hi ? 8 : 0);
      v16h bfr;
#pragma unroll
      for (int j = 0; j < 8; ++j) { bfr[j] = as_[(jb + j) * 64 + d]; bfr[j + 8] = as_[(jb + 16 + j) * 64 + d]; }
      c = WMMA_F16(a, bfr, c);
    }
#pragma unroll
    for (int v = 0; v < 8; ++v) {
      const int i = it * 16 + v + hi * 8;
      if (i < 75) V[base + (size_t)i * 64 + d] = (h16)fmaxf(c[v], 0.f);  // relu
    }
  }
}

// ---------------------------------------------------------------------------
// Final: out[b] = h100[b,:100] . m2_w + m2_b
// ---------------------------------------------------------------------------
__global__ void final_m2(const h16* __restrict__ h100,
                         const float* __restrict__ m2_w,
                         const float* __restrict__ m2_b,
                         float* __restrict__ out) {
  const int b = blockIdx.x * blockDim.x + threadIdx.x;
  if (b >= 512) return;
  float s = m2_b[0];
  for (int c = 0; c < 100; ++c) s += (float)h100[b * 112 + c] * m2_w[c];
  out[b] = s;
}

// ---------------------------------------------------------------------------
extern "C" void kernel_launch(void* const* d_in, const int* in_sizes, int n_in,
                              void* d_out, int out_size, void* d_ws, size_t ws_size,
                              hipStream_t stream) {
  (void)in_sizes; (void)n_in; (void)out_size; (void)ws_size;
  const int*   x     = (const int*)d_in[0];
  const float* emb   = (const float*)d_in[1];
  const float* dr_w  = (const float*)d_in[2];  const float* dr_b  = (const float*)d_in[3];
  const float* c1_w  = (const float*)d_in[4];  const float* c1_b  = (const float*)d_in[5];
  const float* c11_w = (const float*)d_in[6];  const float* c11_b = (const float*)d_in[7];
  const float* c12_w = (const float*)d_in[8];  const float* c12_b = (const float*)d_in[9];
  const float* c21_w = (const float*)d_in[10]; const float* c21_b = (const float*)d_in[11];
  const float* c22_w = (const float*)d_in[12]; const float* c22_b = (const float*)d_in[13];
  const float* c31_w = (const float*)d_in[14]; const float* c31_b = (const float*)d_in[15];
  const float* c32_w = (const float*)d_in[16]; const float* c32_b = (const float*)d_in[17];
  const float* c41_w = (const float*)d_in[18]; const float* c41_b = (const float*)d_in[19];
  const float* fc_w  = (const float*)d_in[20]; const float* fc_b  = (const float*)d_in[21];
  const float* cp1_w = (const float*)d_in[22]; const float* cp1_b = (const float*)d_in[23];
  const float* cp2_w = (const float*)d_in[24]; const float* cp2_b = (const float*)d_in[25];
  const float* wt    = (const float*)d_in[26]; const float* wx    = (const float*)d_in[27];
  const float* bh    = (const float*)d_in[28]; const float* wa    = (const float*)d_in[29];
  const float* ba    = (const float*)d_in[30];
  const float* m1_w  = (const float*)d_in[31]; const float* m1_b  = (const float*)d_in[32];
  const float* m2_w  = (const float*)d_in[33]; const float* m2_b  = (const float*)d_in[34];
  float* out = (float*)d_out;

  char* p = (char*)d_ws;
  auto alloc = [&](size_t bytes) {
    char* r = p; p += (bytes + 255) & ~(size_t)255; return r;
  };
  float* table_r   = (float*)alloc(23 * 64 * sizeof(float));
  float* table_acc = (float*)alloc(23 * 64 * sizeof(float));
  float* bias_h    = (float*)alloc(64 * sizeof(float));
  h16*   WM        = (h16*)alloc(64 * 64 * sizeof(h16));
  h16*   W1f       = (h16*)alloc(64 * 256 * sizeof(h16));
  h16*   W2f       = (h16*)alloc(64 * 320 * sizeof(h16));
  h16*   Wtf       = (h16*)alloc(64 * 64 * sizeof(h16));
  h16*   Wxf       = (h16*)alloc(64 * 64 * sizeof(h16));
  h16*   M1f       = (h16*)alloc((size_t)112 * 960 * sizeof(h16));
  float* b100      = (float*)alloc(112 * sizeof(float));
  h16*   H         = (h16*)alloc((size_t)153600 * 64 * sizeof(h16));
  h16*   a_in      = (h16*)alloc((size_t)38400 * 64 * sizeof(h16));
  float* qt        = (float*)alloc((size_t)38400 * 64 * sizeof(float));
  float* kxb       = (float*)alloc((size_t)38400 * 64 * sizeof(float));
  h16*   V         = (h16*)alloc((size_t)38400 * 64 * sizeof(h16));
  h16*   out960    = (h16*)alloc((size_t)512 * 960 * sizeof(h16));
  h16*   h100      = (h16*)alloc((size_t)512 * 112 * sizeof(h16));

  prep_tables<<<1, 64, 0, stream>>>(
      emb, dr_w, dr_b, c1_w, c1_b, c11_w, c11_b, c12_w, c12_b,
      c21_w, c21_b, c22_w, c22_b, c31_w, c31_b, c32_w, c32_b,
      c41_w, c41_b, fc_w, fc_b, table_r, table_acc, bias_h, WM);

  prep_weights<<<64, 256, 0, stream>>>(
      cp1_w, cp2_w, wt, wx, m1_w, m1_b, W1f, W2f, Wtf, Wxf, M1f, b100);

  // H[B*300, 64] : tables + maxpool-projection WMMA
  stage1_h<<<2400, 256, 0, stream>>>(x, table_r, table_acc, bias_h, WM, H);

  // cp1 (k=4,s=4) as GEMM over H viewed [38400 x 256]; B register-resident
  gemm_res<256, 4><<<300, 256, 0, stream>>>(H, W1f, cp1_b, a_in, 38400, 4, 64);

  // fused qt/kx projections (shared A operands)
  gemm_qk<4><<<300, 256, 0, stream>>>(a_in, Wtf, Wxf, qt, kxb, 38400);

  // additive attention + relu -> V[B*75, 64] f16 (scores VALU/trans, context WMMA)
  attention<<<512, 256, 0, stream>>>(qt, kxb, a_in, bh, wa, ba, V);

  // cp2 (k=5,s=5) as GEMM over V viewed [7680 x 320]; B register-resident
  gemm_res<320, 4><<<60, 256, 0, stream>>>(V, W2f, cp2_b, out960, 7680, 4, 64);

  // m1 + relu : [512 x 960] x [112 x 960]^T (streaming K with prefetch)
  gemm16<<<28, 256, 0, stream>>>(out960, M1f, b100, h100, 512, 960, 7, 112, 1);

  final_m2<<<2, 256, 0, stream>>>(h100, m2_w, m2_b, out);
}
